// MHA_SP2NormEager_84559316123909
// MI455X (gfx1250) — matrix operand order repaired and checked by reading, not compiled, not run on hardware
//
#include <hip/hip_runtime.h>

#define B_   2
#define T_   2048
#define C_   1024
#define H_   16
#define HS_  64
#define M_   (B_*T_)      // 4096 rows of x
#define NQKV (3*C_)       // 3072

typedef __bf16 bf16;
typedef __attribute__((ext_vector_type(16))) __bf16 v16bf;
typedef __attribute__((ext_vector_type(8)))  __bf16 v8bf;
typedef __attribute__((ext_vector_type(8)))  float  v8f;

__device__ __forceinline__ v16bf join16(v8bf lo, v8bf hi) {
  v16bf r;
#pragma unroll
  for (int i = 0; i < 8; ++i) { r[i] = lo[i]; r[8 + i] = hi[i]; }
  return r;
}

__device__ __forceinline__ v8f wmma_bf16(v16bf a, v16bf b, v8f c) {
  // D = A(16x32 bf16) * B(32x16 bf16) + C(16x16 f32)
  return __builtin_amdgcn_wmma_f32_16x16x32_bf16(
      /*neg_a=*/false, a, /*neg_b=*/false, b,
      /*c_mod=*/(short)0, c, /*reuse_a=*/false, /*reuse_b=*/false);
}

// ---------------------------------------------------------------------------
// fp32 -> bf16 conversion, 4 elements / thread (b128 load, b64 store)
// ---------------------------------------------------------------------------
__global__ void cvt_f32_bf16_x4(const float* __restrict__ s, bf16* __restrict__ d,
                                int n4) {
  int i = blockIdx.x * blockDim.x + threadIdx.x;
  if (i < n4) {
    float4 f = ((const float4*)s)[i];
    union { bf16 h[4]; unsigned long long u; } o;
    o.h[0] = (bf16)f.x; o.h[1] = (bf16)f.y;
    o.h[2] = (bf16)f.z; o.h[3] = (bf16)f.w;
    ((unsigned long long*)d)[i] = o.u;
  }
}

// ---------------------------------------------------------------------------
// QKV projection: qkv[m,f] = sum_k X[m,k] * Wqkv[f,k]
// One wave -> 32x64 output tile (2 A-frags x 4 B-frags -> 8 WMMA / k-step).
// Scatter epilogue into Q[B,H,T,HS], K[B,H,T,HS], V^T[B,H,HS,T] (bf16).
// grid: 128 M-tiles x 48 N-tiles = 6144 waves.
// ---------------------------------------------------------------------------
__global__ void qkv_gemm(const bf16* __restrict__ X, const bf16* __restrict__ W,
                         bf16* __restrict__ Q, bf16* __restrict__ Kc,
                         bf16* __restrict__ Vt) {
  const int lane = threadIdx.x & 31;
  const int wid  = blockIdx.x * (blockDim.x >> 5) + (threadIdx.x >> 5);
  const int tm = wid & 127;                 // M_/32 = 128
  const int tn = wid >> 7;                  // NQKV/64 = 48
  const int m0 = tm * 32, n0 = tn * 64;
  const int hlf = lane >> 4, lc = lane & 15;

  const bf16* xrow0 = X + (size_t)(m0 + lc) * C_;
  const bf16* xrow1 = X + (size_t)(m0 + 16 + lc) * C_;
  v8f acc[2][4] = {};
  for (int ks = 0; ks < C_; ks += 32) {
    v16bf a0 = join16(*(const v8bf*)(xrow0 + ks + hlf * 16),
                      *(const v8bf*)(xrow0 + ks + 8 + hlf * 16));
    v16bf a1 = join16(*(const v8bf*)(xrow1 + ks + hlf * 16),
                      *(const v8bf*)(xrow1 + ks + 8 + hlf * 16));
#pragma unroll
    for (int nf = 0; nf < 4; ++nf) {
      const bf16* wrow = W + (size_t)(n0 + nf * 16 + lc) * C_;
      v16bf b = *(const v16bf*)(wrow + ks + hlf * 16);
      acc[0][nf] = wmma_bf16(a0, b, acc[0][nf]);
      acc[1][nf] = wmma_bf16(a1, b, acc[1][nf]);
    }
  }
#pragma unroll
  for (int mi = 0; mi < 2; ++mi) {
#pragma unroll
    for (int nf = 0; nf < 4; ++nf) {
#pragma unroll
      for (int g = 0; g < 8; ++g) {
        int m = m0 + mi * 16 + g + hlf * 8;
        int f = n0 + nf * 16 + lc;
        int which = f / C_;                 // 0:q 1:k 2:v
        int cidx  = f % C_;
        int h = cidx / HS_, hs = cidx % HS_;
        int bb = m / T_, tt = m % T_;
        bf16 bv = (bf16)acc[mi][nf][g];
        size_t bh = (size_t)(bb * H_ + h);
        if (which == 0)      Q [(bh * T_ + tt) * HS_ + hs] = bv;
        else if (which == 1) Kc[(bh * T_ + tt) * HS_ + hs] = bv;
        else                 Vt[(bh * HS_ + hs) * T_ + tt] = bv;
      }
    }
  }
}

// ---------------------------------------------------------------------------
// Attention, sp2 (squared softplus), causal, eager row-normalization.
// One wave per (b,h, 32-row q-pair): two 16-row q-tiles share all K/V
// fragments. Single pass: O_un = sum_j P_j V_j ; rs = rowsum(P) ;
// O = O_un / (rs + 1e-9). P staged via LDS (K=32 A-fragment repack).
// qt1 = 2*qp+1 is odd, so kv pairs never overrun: only tile0's last P half
// needs zero fill.
// ---------------------------------------------------------------------------
__global__ void attn_sp2(const bf16* __restrict__ Q, const bf16* __restrict__ Kc,
                         const bf16* __restrict__ Vt, bf16* __restrict__ O) {
  __shared__ __align__(16) bf16 Pl[2][16][32];
  const int lane = threadIdx.x & 31;
  const int hlf = lane >> 4, lc = lane & 15;
  const int nqp = T_ / 32;                  // 64 q-pairs
  const int qp = blockIdx.x % nqp;
  const int bh = blockIdx.x / nqp;          // 0..B*H-1
  const int bidx = bh / H_, hidx = bh % H_;
  const int qt0 = qp * 2, qt1 = qp * 2 + 1;

  const bf16* Qh = Q  + (size_t)bh * T_ * HS_;
  const bf16* Kh = Kc + (size_t)bh * T_ * HS_;
  const bf16* Vh = Vt + (size_t)bh * HS_ * T_;

  // Q A-fragments for both q-tiles (K-dim = HS = 64 = 2 x 32)
  v16bf aq[2][2];
#pragma unroll
  for (int t = 0; t < 2; ++t) {
    const bf16* qrow = Qh + (size_t)((qt0 + t) * 16 + lc) * HS_;
    aq[t][0] = join16(*(const v8bf*)(qrow + hlf * 16),
                      *(const v8bf*)(qrow + 8 + hlf * 16));
    aq[t][1] = join16(*(const v8bf*)(qrow + 32 + hlf * 16),
                      *(const v8bf*)(qrow + 40 + hlf * 16));
  }

  v8f o[2][4] = {};
  float rs[2][8] = {};

  for (int j0 = 0; j0 <= qt1; j0 += 2) {
    // prefetch next kv-pair's K rows toward the WGP while we compute
    if (j0 + 2 <= qt1) {
      __builtin_prefetch(Kh + (size_t)((j0 + 2) * 16 + lc) * HS_, 0, 1);
      __builtin_prefetch(Vh + (size_t)lc * T_ + (j0 + 2) * 16, 0, 1);
    }
#pragma unroll
    for (int jj = 0; jj < 2; ++jj) {
      const int j = j0 + jj;                // j <= qt1 always (qt1 odd)
      const bf16* krow = Kh + (size_t)(j * 16 + lc) * HS_;
      v16bf b0 = *(const v16bf*)(krow + hlf * 16);
      v16bf b1 = *(const v16bf*)(krow + 32 + hlf * 16);
#pragma unroll
      for (int t = 0; t < 2; ++t) {
        const int qt = qt0 + t;
        if (j <= qt) {
          v8f s = {};
          s = wmma_bf16(aq[t][0], b0, s);
          s = wmma_bf16(aq[t][1], b1, s);
#pragma unroll
          for (int g = 0; g < 8; ++g) {
            float x = s[g] * 0.125f;                    // hs^-0.5
            float sp = (x > 20.f) ? x : log1pf(__expf(x));
            float p = sp * sp;
            if (j == qt && lc > g + hlf * 8) p = 0.f;   // causal diag mask
            rs[t][g] += p;
            Pl[t][g + hlf * 8][jj * 16 + lc] = (bf16)p;
          }
        } else {  // only tile0's jj=1 at the final pair
#pragma unroll
          for (int g = 0; g < 8; ++g)
            Pl[t][g + hlf * 8][jj * 16 + lc] = (bf16)0.f;
        }
      }
    }
    __syncthreads();
    v16bf ap[2];
#pragma unroll
    for (int t = 0; t < 2; ++t)
      ap[t] = join16(*(const v8bf*)(&Pl[t][lc][hlf * 16]),
                     *(const v8bf*)(&Pl[t][lc][8 + hlf * 16]));
#pragma unroll
    for (int nf = 0; nf < 4; ++nf) {
      // V^T row (hs index) contiguous over keys -> contiguous B-fragment
      const bf16* vrow = Vh + (size_t)(nf * 16 + lc) * T_ + j0 * 16;
      v16bf bv = *(const v16bf*)(vrow + hlf * 16);
      o[0][nf] = wmma_bf16(ap[0], bv, o[0][nf]);
      o[1][nf] = wmma_bf16(ap[1], bv, o[1][nf]);
    }
    __syncthreads();
  }

  // reduce row sums across the 16 lanes of each half (wave32)
#pragma unroll
  for (int t = 0; t < 2; ++t) {
#pragma unroll
    for (int g = 0; g < 8; ++g) {
      float r = rs[t][g];
      r += __shfl_xor(r, 1, 32);
      r += __shfl_xor(r, 2, 32);
      r += __shfl_xor(r, 4, 32);
      r += __shfl_xor(r, 8, 32);
      rs[t][g] = r + 1e-9f;
    }
  }

  // O layout: [B, T, C] with c = h*HS + n  (feeds proj GEMM directly)
#pragma unroll
  for (int t = 0; t < 2; ++t) {
#pragma unroll
    for (int nf = 0; nf < 4; ++nf) {
#pragma unroll
      for (int g = 0; g < 8; ++g) {
        int tt = (qt0 + t) * 16 + g + hlf * 8;
        int cc = hidx * HS_ + nf * 16 + lc;
        O[((size_t)bidx * T_ + tt) * C_ + cc] = (bf16)(o[t][nf][g] / rs[t][g]);
      }
    }
  }
}

// ---------------------------------------------------------------------------
// Output projection: Y[m,f] = sum_k A[m,k] * Wproj[f,k] + bias[f]  (fp32 out)
// One wave -> 32x64 output tile. grid: 128 x 16 wave-tiles.
// ---------------------------------------------------------------------------
__global__ void proj_gemm(const bf16* __restrict__ A, const bf16* __restrict__ W,
                          const float* __restrict__ bias, float* __restrict__ Y) {
  const int lane = threadIdx.x & 31;
  const int wid  = blockIdx.x * (blockDim.x >> 5) + (threadIdx.x >> 5);
  const int tm = wid & 127;                 // 128 M-tiles
  const int tn = wid >> 7;                  // 16 N-tiles
  const int m0 = tm * 32, n0 = tn * 64;
  const int hlf = lane >> 4, lc = lane & 15;

  const bf16* arow0 = A + (size_t)(m0 + lc) * C_;
  const bf16* arow1 = A + (size_t)(m0 + 16 + lc) * C_;
  v8f acc[2][4] = {};
  for (int ks = 0; ks < C_; ks += 32) {
    v16bf a0 = join16(*(const v8bf*)(arow0 + ks + hlf * 16),
                      *(const v8bf*)(arow0 + ks + 8 + hlf * 16));
    v16bf a1 = join16(*(const v8bf*)(arow1 + ks + hlf * 16),
                      *(const v8bf*)(arow1 + ks + 8 + hlf * 16));
#pragma unroll
    for (int nf = 0; nf < 4; ++nf) {
      const bf16* wrow = W + (size_t)(n0 + nf * 16 + lc) * C_;
      v16bf b = *(const v16bf*)(wrow + ks + hlf * 16);
      acc[0][nf] = wmma_bf16(a0, b, acc[0][nf]);
      acc[1][nf] = wmma_bf16(a1, b, acc[1][nf]);
    }
  }
#pragma unroll
  for (int mi = 0; mi < 2; ++mi) {
#pragma unroll
    for (int nf = 0; nf < 4; ++nf) {
#pragma unroll
      for (int g = 0; g < 8; ++g) {
        int m = m0 + mi * 16 + g + hlf * 8;
        int f = n0 + nf * 16 + lc;
        Y[(size_t)m * C_ + f] = acc[mi][nf][g] + bias[f];
      }
    }
  }
}

// ---------------------------------------------------------------------------
extern "C" void kernel_launch(void* const* d_in, const int* in_sizes, int n_in,
                              void* d_out, int out_size, void* d_ws, size_t ws_size,
                              hipStream_t stream) {
  const float* x      = (const float*)d_in[0];
  const float* w_qkv  = (const float*)d_in[1];
  const float* w_proj = (const float*)d_in[2];
  const float* b_proj = (const float*)d_in[3];
  float* out = (float*)d_out;

  char* ws = (char*)d_ws;
  size_t off = 0;
  auto alloc = [&](size_t bytes) -> void* {
    void* p = ws + off;
    off += (bytes + 255) & ~(size_t)255;
    return p;
  };
  bf16* Xb  = (bf16*)alloc((size_t)M_ * C_ * sizeof(bf16));     // 8 MB
  bf16* Wqb = (bf16*)alloc((size_t)NQKV * C_ * sizeof(bf16));   // 6 MB
  bf16* Wpb = (bf16*)alloc((size_t)C_ * C_ * sizeof(bf16));     // 2 MB
  bf16* Qb  = (bf16*)alloc((size_t)M_ * C_ * sizeof(bf16));     // 8 MB
  bf16* Kb  = (bf16*)alloc((size_t)M_ * C_ * sizeof(bf16));     // 8 MB
  bf16* Vtb = (bf16*)alloc((size_t)M_ * C_ * sizeof(bf16));     // 8 MB
  bf16* Ob  = Xb;  // X is consumed by qkv_gemm before attn writes here

  int n1 = (M_ * C_) / 4;
  int n2 = (NQKV * C_) / 4;
  int n3 = (C_ * C_) / 4;
  cvt_f32_bf16_x4<<<(n1 + 255) / 256, 256, 0, stream>>>(x, Xb, n1);
  cvt_f32_bf16_x4<<<(n2 + 255) / 256, 256, 0, stream>>>(w_qkv, Wqb, n2);
  cvt_f32_bf16_x4<<<(n3 + 255) / 256, 256, 0, stream>>>(w_proj, Wpb, n3);

  // QKV: 128 x 48 wave-tiles, 8 waves per 256-thread block
  qkv_gemm<<<(128 * 48) / 8, 256, 0, stream>>>(Xb, Wqb, Qb, Kb, Vtb);

  // Attention: one wave per (b,h,q-pair): 32 * 64 = 2048 blocks
  attn_sp2<<<B_ * H_ * (T_ / 32), 32, 0, stream>>>(Qb, Kb, Vtb, Ob);

  // Output projection: 128 x 16 wave-tiles
  proj_gemm<<<(128 * 16) / 8, 256, 0, stream>>>(Ob, Wpb, b_proj, out);
}
